// GraphConvLayer_68831145886533
// MI455X (gfx1250) — matrix operand order repaired and checked by reading, not compiled, not run on hardware
//
#include <hip/hip_runtime.h>

typedef __attribute__((ext_vector_type(16))) _Float16     v16h;
typedef __attribute__((ext_vector_type(8)))  _Float16     v8h;
typedef __attribute__((ext_vector_type(8)))  float        v8f;
typedef __attribute__((ext_vector_type(4)))  unsigned int v4u;
typedef __attribute__((ext_vector_type(8)))  int          v8i;
typedef __attribute__((ext_vector_type(4)))  int          v4i;

#define N_NODES 8192
#define IN_F    128
#define OUT_F   128

#if defined(__has_builtin)
#if __has_builtin(__builtin_amdgcn_tensor_load_to_lds) && \
    __has_builtin(__builtin_amdgcn_s_wait_tensorcnt)
#define HAVE_TDM 1
#endif
#endif

// ---------------------------------------------------------------------------
// Kernel 1: per-row softmax statistics.  One block per row, 256 threads.
// ---------------------------------------------------------------------------
__global__ __launch_bounds__(256)
void row_stats_kernel(const float* __restrict__ A,
                      float* __restrict__ row_max,
                      float* __restrict__ row_rsum)
{
    const int row = blockIdx.x;
    const int t   = threadIdx.x;
    const float4* rp = (const float4*)(A + (size_t)row * N_NODES);

    float4 buf[8];
#pragma unroll
    for (int i = 0; i < 8; ++i) buf[i] = rp[i * 256 + t];

    float m = -INFINITY;
#pragma unroll
    for (int i = 0; i < 8; ++i) {
        m = fmaxf(m, fmaxf(fmaxf(buf[i].x, buf[i].y), fmaxf(buf[i].z, buf[i].w)));
    }
    float s = 0.f;
#pragma unroll
    for (int i = 0; i < 8; ++i) {
        s += __expf(buf[i].x - m) + __expf(buf[i].y - m) +
             __expf(buf[i].z - m) + __expf(buf[i].w - m);
    }

    __shared__ float sm[256];
    __shared__ float ss[256];
    sm[t] = m; ss[t] = s;
    __syncthreads();
#pragma unroll
    for (int off = 128; off > 0; off >>= 1) {
        if (t < off) {
            float m2 = sm[t + off], s2 = ss[t + off];
            float m1 = sm[t],       s1 = ss[t];
            float mm = fmaxf(m1, m2);
            ss[t] = s1 * __expf(m1 - mm) + s2 * __expf(m2 - mm);
            sm[t] = mm;
        }
        __syncthreads();
    }
    if (t == 0) {
        row_max[row]  = sm[0];
        row_rsum[row] = 1.0f / ss[0];
    }
}

// ---------------------------------------------------------------------------
// Kernel 1b: XT[c][k] = (f16) X[k][c]   (2MB, one-time; feeds the TDM)
// blockIdx.x = c; packed b32 stores, fully coalesced on the write side.
// ---------------------------------------------------------------------------
union H2U { _Float16 h[2]; unsigned int u; };

__global__ __launch_bounds__(256)
void xt_prep_kernel(const float* __restrict__ X, _Float16* __restrict__ XT)
{
    const int c = blockIdx.x;                 // 0..127
    const int t = threadIdx.x;                // 0..255
    unsigned int* dst = (unsigned int*)(XT + (size_t)c * N_NODES);
#pragma unroll
    for (int i = 0; i < 16; ++i) {
        int k = (i * 256 + t) * 2;
        float a = X[(size_t)k * IN_F + c];
        float b = X[(size_t)(k + 1) * IN_F + c];
        H2U v; v.h[0] = (_Float16)a; v.h[1] = (_Float16)b;
        dst[k >> 1] = v.u;
    }
}

// ---------------------------------------------------------------------------
// Helpers
// ---------------------------------------------------------------------------
__device__ inline v16h pack16(v8h lo, v8h hi) {
    v16h r;
#pragma unroll
    for (int i = 0; i < 8; ++i) { r[i] = lo[i]; r[i + 8] = hi[i]; }
    return r;
}

#ifdef HAVE_TDM
// TDM: copy the 2-D tile XT[0..127][kc..kc+31] (f16) into LDS at lds_off.
// D# per CDNA5 ISA ch.8: group0 = {count/flags, lds_addr, global_addr, type},
// group1 = {mask/data_size, dims/tile dims, dim0 stride}.  2-D tensor ->
// groups 2/3 zero.  LDS fill order: x (k, 32 halves) fastest, then y (c),
// which reproduces the [c][k] layout the B-operand ds_load_b128s expect.
__device__ inline void tdm_load_chunk(unsigned lds_off, const _Float16* gptr)
{
    unsigned long long ga = (unsigned long long)(size_t)gptr;
    v4u g0;
    g0[0] = 1u;                                      // count=1, user descriptor
    g0[1] = lds_off;                                 // lds_addr (bytes)
    g0[2] = (unsigned)(ga & 0xFFFFFFFFu);            // global_addr[31:0]
    g0[3] = (unsigned)((ga >> 32) & 0x01FFFFFFu)     // global_addr[56:32]
          | (2u << 30);                              // type = 2 ("image")
    v8i g1;
    g1[0] = 1 << 16;                                 // data_size = 1 (2 bytes)
    g1[1] = (int)(8192u << 16);                      // tensor_dim0 = 8192 (lo16)
    g1[2] = (int)(128u << 16);                       // dim0 hi=0 | tensor_dim1=128
    g1[3] = (int)(32u << 16);                        // dim1 hi=0 | tile_dim0 = 32
    g1[4] = 128;                                     // tile_dim1 = 128, tile_dim2=0
    g1[5] = 8192;                                    // tensor_dim0_stride lo32
    g1[6] = 0;                                       // stride hi | dim1_stride lo
    g1[7] = 0;
    v4i z4 = {0, 0, 0, 0};
#if __clang_major__ >= 23
    v8i z8 = {0, 0, 0, 0, 0, 0, 0, 0};
    __builtin_amdgcn_tensor_load_to_lds(g0, g1, z4, z4, z8, 0);
#else
    __builtin_amdgcn_tensor_load_to_lds(g0, g1, z4, z4, 0);
#endif
}
#endif

// ---------------------------------------------------------------------------
// Kernel 2: fused  out = relu( softmax(A) @ X @ W^T + b )
// 128 blocks x 128 threads (4 wave32).  Wave w owns rows
// [blockIdx*64 + 16w, +16).  All matrix math via v_wmma_f32_16x16x32_f16.
// X chunks are streamed into LDS by the Tensor Data Mover (wave 0 programs
// the D#, s_wait_tensorcnt + block barrier publishes the buffer).
//
// CDNA5 A-operand map (16-bit, 16x32):  lane = M (mod 16), h = lane/16,
//   elems 0..7 -> K = 8h+e, elems 8..15 -> K = 8h+16+(e-8).
// CDNA5 B-operand map (16-bit, 32x16):  lane = N (mod 16), h = lane/16,
//   elems 0..7 -> K = 8h+e, elems 8..15 -> K = 8h+16+(e-8).
// C/D map: lane = N (mod 16), VGPR r -> M = r + 8h.
// ---------------------------------------------------------------------------
__global__ __launch_bounds__(128, 1)
void gcn_fused_kernel(const float* __restrict__ A,
                      const _Float16* __restrict__ XT,   // [128][8192] f16
                      const float* __restrict__ W,
                      const float* __restrict__ bias,
                      const float* __restrict__ row_max,
                      const float* __restrict__ row_rsum,
                      float* __restrict__ out)
{
    __shared__ _Float16 ldsX[2][IN_F * 32];     // [c][k] chunk, 8KB each
    __shared__ _Float16 ldsAgg[4][16 * IN_F];   // per-wave agg tile, 4KB each

    const int tid  = threadIdx.x;
    const int wave = tid >> 5;
    const int lane = tid & 31;
    const int lm   = lane & 15;   // M (A) or N (B, C/D) within tile
    const int h    = lane >> 4;   // half-wave select

    const int row0 = blockIdx.x * 64 + wave * 16;   // wave's first row
    const int arow_i = row0 + lm;                   // this lane's A row
    const float rmax = row_max[arow_i];
    const float rrs  = row_rsum[arow_i];
    const float* __restrict__ arow = A + (size_t)arow_i * N_NODES;

    const unsigned ldsx_off[2] = {
        (unsigned)(size_t)&ldsX[0][0],
        (unsigned)(size_t)&ldsX[1][0]
    };

    const v8f vzero = {0.f, 0.f, 0.f, 0.f, 0.f, 0.f, 0.f, 0.f};
    v8f acc[8];
#pragma unroll
    for (int t = 0; t < 8; ++t) acc[t] = vzero;

    // ---- prologue: stage chunk 0 ----
#ifdef HAVE_TDM
    if (wave == 0) {
        tdm_load_chunk(ldsx_off[0], XT);
        __builtin_amdgcn_s_wait_tensorcnt(0);
    }
#else
    {   // fallback: thread t copies column c = t (32 halves = 4 x b128)
        const uint4* src = (const uint4*)(XT + (size_t)tid * N_NODES);
        uint4* d = (uint4*)(&ldsX[0][tid * 32]);
#pragma unroll
        for (int i = 0; i < 4; ++i) d[i] = src[i];
    }
#endif
    __syncthreads();

    for (int kc = 0; kc < N_NODES; kc += 32) {
        const int cur = (kc >> 5) & 1;
        const bool more = (kc + 32 < N_NODES);
        if (more) {
#ifdef HAVE_TDM
            if (wave == 0)   // TDM streams next chunk while we compute
                tdm_load_chunk(ldsx_off[cur ^ 1], XT + (kc + 32));
#else
            const uint4* src = (const uint4*)(XT + (size_t)tid * N_NODES + kc + 32);
            uint4* d = (uint4*)(&ldsX[cur ^ 1][tid * 32]);
#pragma unroll
            for (int i = 0; i < 4; ++i) d[i] = src[i];
#endif
            // pull next adjacency segment (one full 128B line per lane)
            __builtin_prefetch(arow + kc + 32 + 8 * h, 0, 0);
        }

        // ---- preload all 8 B tiles from LDS ----
        v16h bm[8];
#pragma unroll
        for (int t = 0; t < 8; ++t) {
            const _Float16* bp = &ldsX[cur][(16 * t + lm) * 32 + 8 * h];
            v8h b0 = *(const v8h*)(bp);
            v8h b1 = *(const v8h*)(bp + 16);
            bm[t] = pack16(b0, b1);
        }

        // ---- A operand: softmax probabilities of this lane's row ----
        const float* ap = arow + kc + 8 * h;
        float4 a0 = *(const float4*)(ap);
        float4 a1 = *(const float4*)(ap + 4);
        float4 a2 = *(const float4*)(ap + 16);
        float4 a3 = *(const float4*)(ap + 20);
        float av[16] = {a0.x, a0.y, a0.z, a0.w, a1.x, a1.y, a1.z, a1.w,
                        a2.x, a2.y, a2.z, a2.w, a3.x, a3.y, a3.z, a3.w};
        v16h pa;
#pragma unroll
        for (int i = 0; i < 16; ++i)
            pa[i] = (_Float16)(__expf(av[i] - rmax) * rrs);

        // ---- 8 back-to-back WMMAs over the N-tiles of X ----
#pragma unroll
        for (int t = 0; t < 8; ++t) {
            acc[t] = __builtin_amdgcn_wmma_f32_16x16x32_f16(
                false, pa, false, bm[t], (short)0, acc[t], false, false);
        }

#ifdef HAVE_TDM
        if (more && wave == 0)
            __builtin_amdgcn_s_wait_tensorcnt(0);   // next chunk landed in LDS
#endif
        __syncthreads();   // publish next buffer / retire current reads
    }

    // ---- re-layout agg (C/D layout) -> A-operand layout via LDS ----
    _Float16* agg = ldsAgg[wave];   // wave-private; same-wave DS ops are in order
#pragma unroll
    for (int t = 0; t < 8; ++t)
#pragma unroll
        for (int r = 0; r < 8; ++r)
            agg[(r + 8 * h) * IN_F + 16 * t + lm] = (_Float16)acc[t][r];

    // ---- second GEMM: out_tile[16 x 128] = agg[16 x 128] @ W^T ----
    v8f oac[8];
#pragma unroll
    for (int t = 0; t < 8; ++t) oac[t] = vzero;

#pragma unroll
    for (int kc2 = 0; kc2 < IN_F; kc2 += 32) {
        const _Float16* ap2 = agg + lm * IN_F + kc2 + 16 * h;
        v8h l0 = *(const v8h*)(ap2);
        v8h l1 = *(const v8h*)(ap2 + 8);
        v16h a2v = pack16(l0, l1);
#pragma unroll
        for (int t = 0; t < 8; ++t) {
            const float* wr = W + (size_t)(16 * t + lm) * IN_F + kc2 + 16 * h;
            float4 w0 = *(const float4*)(wr);
            float4 w1 = *(const float4*)(wr + 4);
            float4 w2 = *(const float4*)(wr + 8);
            float4 w3 = *(const float4*)(wr + 12);
            float wv[16] = {w0.x, w0.y, w0.z, w0.w, w1.x, w1.y, w1.z, w1.w,
                            w2.x, w2.y, w2.z, w2.w, w3.x, w3.y, w3.z, w3.w};
            v16h b2v;
#pragma unroll
            for (int i = 0; i < 16; ++i) b2v[i] = (_Float16)wv[i];
            oac[t] = __builtin_amdgcn_wmma_f32_16x16x32_f16(
                false, a2v, false, b2v, (short)0, oac[t], false, false);
        }
    }

    // ---- bias + relu + store (f32) ----
#pragma unroll
    for (int t = 0; t < 8; ++t) {
        const float bv = bias[16 * t + lm];
#pragma unroll
        for (int r = 0; r < 8; ++r) {
            const int row = row0 + r + 8 * h;
            float v = oac[t][r] + bv;
            out[(size_t)row * OUT_F + 16 * t + lm] = v > 0.f ? v : 0.f;
        }
    }
}

// ---------------------------------------------------------------------------
// Launch
// ---------------------------------------------------------------------------
extern "C" void kernel_launch(void* const* d_in, const int* in_sizes, int n_in,
                              void* d_out, int out_size, void* d_ws, size_t ws_size,
                              hipStream_t stream)
{
    const float* A  = (const float*)d_in[0];   // [8192, 8192]
    const float* X  = (const float*)d_in[1];   // [8192, 128]
    const float* W  = (const float*)d_in[2];   // [128, 128]
    const float* b  = (const float*)d_in[3];   // [128]
    float* out      = (float*)d_out;           // [8192, 128]

    float*    row_max  = (float*)d_ws;                    // 8192 f32
    float*    row_rsum = row_max + N_NODES;               // 8192 f32
    _Float16* XT       = (_Float16*)(row_rsum + N_NODES); // [128][8192] f16, 2MB

    row_stats_kernel<<<N_NODES, 256, 0, stream>>>(A, row_max, row_rsum);
    xt_prep_kernel<<<IN_F, 256, 0, stream>>>(X, XT);
    gcn_fused_kernel<<<N_NODES / 64, 128, 0, stream>>>(A, XT, W, b,
                                                       row_max, row_rsum, out);
}
